// CrossAttention_9268539425271
// MI455X (gfx1250) — matrix operand order repaired and checked by reading, not compiled, not run on hardware
//
#include <hip/hip_runtime.h>
#include <hip/hip_bf16.h>

// B=4, C=256, N=4096, OUT=256. temp = sqrt(256) = 16.
//   proj_kernel: Q[b][n][o], K[b][n][o], Vt[b][o][m] in bf16 (workspace), via
//                v_wmma_f32_16x16x32_bf16 with W chunks staged in LDS.
//   attn_kernel: flash attention; K/V tiles streamed HBM->LDS with the Tensor
//                Data Mover (double-buffered, TENSORcnt), online softmax,
//                P transposed through LDS, output (B, OUT, N) fp32.

#define DEVI __device__ __forceinline__

typedef __attribute__((ext_vector_type(16))) __bf16 bf16x16;
typedef __attribute__((ext_vector_type(8)))  __bf16 bf16x8;
typedef __attribute__((ext_vector_type(8)))  float  v8f;
typedef __attribute__((ext_vector_type(4)))  unsigned int v4u;
typedef __attribute__((ext_vector_type(8)))  int v8i;
typedef __attribute__((ext_vector_type(4)))  int v4i;

constexpr int BB  = 4;
constexpr int CC  = 256;
constexpr int NN  = 4096;
constexpr int OC  = 256;

DEVI v8f wmma_bf16(bf16x16 a, bf16x16 b, v8f c) {
  // D = A(16x32) * B(32x16) + C, fp32 accumulate
  return __builtin_amdgcn_wmma_f32_16x16x32_bf16(false, a, false, b, (short)0, c,
                                                 false, false);
}

DEVI v8f vzero8() {
  v8f z;
#pragma unroll
  for (int i = 0; i < 8; ++i) z[i] = 0.0f;
  return z;
}

// B-operand: lane holds 16 contiguous K values (32B), 16B-aligned -> 2x b128
DEVI bf16x16 ld_g16(const __bf16* p) {
  bf16x8 lo = *(const bf16x8*)(p);
  bf16x8 hi = *(const bf16x8*)(p + 8);
  bf16x16 r;
#pragma unroll
  for (int i = 0; i < 8; ++i) { r[i] = lo[i]; r[i + 8] = hi[i]; }
  return r;
}

// A-operand: lane holds K = {base+0..7, base+16..23} (two 16B runs)
DEVI bf16x16 ld_g_ab(const __bf16* p) {
  bf16x8 lo = *(const bf16x8*)(p);
  bf16x8 hi = *(const bf16x8*)(p + 16);
  bf16x16 r;
#pragma unroll
  for (int i = 0; i < 8; ++i) { r[i] = lo[i]; r[i + 8] = hi[i]; }
  return r;
}

// element-wise variants for (possibly) unaligned LDS rows
DEVI bf16x16 ld16_elem(const __bf16* p) {
  bf16x16 r;
#pragma unroll
  for (int i = 0; i < 16; ++i) r[i] = p[i];
  return r;
}
DEVI bf16x16 ld_ab_elem(const __bf16* p) {
  bf16x16 r;
#pragma unroll
  for (int i = 0; i < 8; ++i) { r[i] = p[i]; r[i + 8] = p[i + 16]; }
  return r;
}

// fp32 global, strided over K (cross-lane coalesced), converted to bf16
DEVI bf16x16 ld_ab_stride_f32(const float* p, int stride) {
  bf16x16 r;
#pragma unroll
  for (int i = 0; i < 8; ++i) {
    r[i]     = (__bf16)p[(size_t)i * stride];
    r[i + 8] = (__bf16)p[(size_t)(i + 16) * stride];
  }
  return r;
}
DEVI bf16x16 ld_col_f32(const float* p, int stride) {
  bf16x16 r;
#pragma unroll
  for (int i = 0; i < 16; ++i) r[i] = (__bf16)p[(size_t)i * stride];
  return r;
}

// ---------------------------------------------------------------------------
// Tensor Data Mover: async 2D tile load HBM -> LDS (D# built per ISA ch.8).
//   data_size = 2 bytes (bf16); LDS row padding via pad_interval/pad_amount
//   to keep ds_load_b128 operand fetches bank-conflict free.
// ---------------------------------------------------------------------------
DEVI void tdm_load_2d(unsigned lds_off, const void* gptr,
                      unsigned dim0, unsigned dim1,
                      unsigned tile0, unsigned tile1,
                      unsigned long long stride0,
                      unsigned pad_interval, unsigned pad_amount) {
  const unsigned long long ga = (unsigned long long)(size_t)gptr;
  v4u g0;
  g0[0] = 1u;  // count=1 valid descriptor, user mode, no gather
  g0[1] = lds_off;
  g0[2] = (unsigned)(ga & 0xffffffffull);
  g0[3] = (unsigned)((ga >> 32) & 0x01ffffffull) | (2u << 30);  // type=2
  v8i g1;
  g1[0] = (int)((1u << 16)                 // data_size = 2 bytes
                | (1u << 20)               // pad_enable
                | (pad_interval << 22)
                | (pad_amount << 25));
  g1[1] = (int)((dim0 & 0xffffu) << 16);                              // tensor_dim0 lo
  g1[2] = (int)(((dim0 >> 16) & 0xffffu) | ((dim1 & 0xffffu) << 16)); // dim0 hi | dim1 lo
  g1[3] = (int)(((dim1 >> 16) & 0xffffu) | ((tile0 & 0xffffu) << 16));
  g1[4] = (int)(tile1 & 0xffffu);                                     // tile_dim1, tile_dim2=0
  g1[5] = (int)(stride0 & 0xffffffffull);
  g1[6] = (int)((stride0 >> 32) & 0xffffull);                         // dim1_stride=0 (2D)
  g1[7] = 0;
  v4i g2 = {0, 0, 0, 0};
  v4i g3 = {0, 0, 0, 0};
#if __clang_major__ >= 23
  v8i g4 = {0, 0, 0, 0, 0, 0, 0, 0};
  __builtin_amdgcn_tensor_load_to_lds(g0, g1, g2, g3, g4, 0);
#else
  __builtin_amdgcn_tensor_load_to_lds(g0, g1, g2, g3, 0);
#endif
}

// ---------------------------------------------------------------------------
// Kernel 1: projections.
//   z=0: Q[b][n][o] = sum_c Wq[o][c] * x[b][c][n]
//   z=1: K[b][n][o] = sum_c Wk[o][c] * xx[b][c][n]
//   z=2: Vt[b][o][m] = sum_c Wv[o][c] * xx[b][c][m]
// grid (N/128, B, 3), block 256 (8 waves)
// ---------------------------------------------------------------------------
__global__ __launch_bounds__(256)
void proj_kernel(const float* __restrict__ x, const float* __restrict__ xx,
                 const float* __restrict__ Wq, const float* __restrict__ Wk,
                 const float* __restrict__ Wv,
                 __bf16* __restrict__ Q, __bf16* __restrict__ K,
                 __bf16* __restrict__ Vt) {
  __shared__ __attribute__((aligned(16))) __bf16 wlds[256 * 34];
  const int tid  = threadIdx.x;
  const int w    = tid >> 5;
  const int lane = tid & 31;
  const int h    = lane >> 4;
  const int l16  = lane & 15;
  const int b    = blockIdx.y;
  const int z    = blockIdx.z;
  const float* Wm = (z == 0) ? Wq : (z == 1) ? Wk : Wv;
  const float* Sm = (z == 0) ? x : xx;

  if (z < 2) {
    __bf16* dst = (z == 0) ? Q : K;
    const int n_base = blockIdx.x * 128 + w * 16;
    v8f acc[16];
#pragma unroll
    for (int i = 0; i < 16; ++i) acc[i] = vzero8();

    for (int cc = 0; cc < 8; ++cc) {
      const int c0 = cc * 32;
      for (int idx = tid; idx < 256 * 32; idx += 256) {
        const int o = idx >> 5, j = idx & 31;
        wlds[o * 34 + j] = (__bf16)Wm[o * 256 + c0 + j];
      }
      __syncthreads();
      const float* ap = Sm + ((size_t)(b * CC + c0 + h * 8)) * NN + (n_base + l16);
      const bf16x16 a = ld_ab_stride_f32(ap, NN);
#pragma unroll
      for (int ot = 0; ot < 16; ++ot) {
        const bf16x16 bm = ld_ab_elem(&wlds[(ot * 16 + l16) * 34 + h * 8]);
        acc[ot] = wmma_bf16(a, bm, acc[ot]);
      }
      __syncthreads();
    }
#pragma unroll
    for (int ot = 0; ot < 16; ++ot)
#pragma unroll
      for (int r = 0; r < 8; ++r)
        dst[((size_t)(b * NN) + n_base + r + 8 * h) * OC + ot * 16 + l16] =
            (__bf16)acc[ot][r];
  } else {
    const int mblk = blockIdx.x * 128;
    v8f acc[2][8];
#pragma unroll
    for (int g = 0; g < 2; ++g)
#pragma unroll
      for (int i = 0; i < 8; ++i) acc[g][i] = vzero8();

    for (int cc = 0; cc < 8; ++cc) {
      const int c0 = cc * 32;
      for (int idx = tid; idx < 256 * 32; idx += 256) {
        const int o = idx >> 5, j = idx & 31;
        wlds[o * 34 + j] = (__bf16)Wm[o * 256 + c0 + j];
      }
      __syncthreads();
      const bf16x16 a0 = ld_ab_elem(&wlds[(w * 16 + l16) * 34 + h * 8]);
      const bf16x16 a1 = ld_ab_elem(&wlds[(128 + w * 16 + l16) * 34 + h * 8]);
#pragma unroll
      for (int mt = 0; mt < 8; ++mt) {
        const float* bp =
            Sm + ((size_t)(b * CC + c0 + h * 8)) * NN + (mblk + mt * 16 + l16);
        const bf16x16 bm = ld_ab_stride_f32(bp, NN);
        acc[0][mt] = wmma_bf16(a0, bm, acc[0][mt]);
        acc[1][mt] = wmma_bf16(a1, bm, acc[1][mt]);
      }
      __syncthreads();
    }
#pragma unroll
    for (int g = 0; g < 2; ++g)
#pragma unroll
      for (int mt = 0; mt < 8; ++mt)
#pragma unroll
        for (int r = 0; r < 8; ++r)
          Vt[((size_t)(b * OC) + g * 128 + w * 16 + r + 8 * h) * NN + mblk +
             mt * 16 + l16] = (__bf16)acc[g][mt][r];
  }
}

// ---------------------------------------------------------------------------
// Kernel 2: flash attention with TDM-streamed K/V tiles.
//   K tile: 64 rows x 256 (row 512B + 16B pad -> 132-dword LDS rows)
//   V tile: 256 rows x 64 (row 128B + 16B pad -> 36-dword LDS rows)
// grid (N/128, B), block 256 (8 waves x 16 query rows)
// ---------------------------------------------------------------------------
constexpr int KROW = 264;  // bf16 elems per padded K row (128+4 dwords)
constexpr int VROW = 72;   // bf16 elems per padded V row (32+4 dwords)

__global__ __launch_bounds__(256)
void attn_kernel(const __bf16* __restrict__ Q, const __bf16* __restrict__ K,
                 const __bf16* __restrict__ Vt, float* __restrict__ out) {
  __shared__ __attribute__((aligned(16))) __bf16 klds[2][64 * KROW];
  __shared__ __attribute__((aligned(16))) __bf16 vlds[2][256 * VROW];
  __shared__ __attribute__((aligned(16))) __bf16 plds[8 * 16 * 72];
  const int tid  = threadIdx.x;
  const int w    = tid >> 5;
  const int lane = tid & 31;
  const int h    = lane >> 4;
  const int l16  = lane & 15;
  const int b    = blockIdx.y;
  const int n_base = blockIdx.x * 128 + w * 16;
  __bf16* pl = plds + w * (16 * 72);

  v8f o_acc[16];
#pragma unroll
  for (int i = 0; i < 16; ++i) o_acc[i] = vzero8();
  float m_run[8], l_run[8];
#pragma unroll
  for (int r = 0; r < 8; ++r) { m_run[r] = -INFINITY; l_run[r] = 0.0f; }

  const __bf16* Qb = Q + ((size_t)(b * NN) + n_base + l16) * OC;
  const __bf16* Kb = K + (size_t)(b * NN) * OC;
  const __bf16* Vb = Vt + (size_t)(b * OC) * NN;

  // prologue: kick off tile 0 into buffer 0
  if (w == 0) {
    tdm_load_2d((unsigned)(size_t)&klds[0][0], Kb, 256u, 4096u, 256u, 64u, 256ull, 6u, 3u);
    tdm_load_2d((unsigned)(size_t)&vlds[0][0], Vb, 4096u, 256u, 64u, 256u, 4096ull, 4u, 3u);
  }

  for (int it = 0; it < NN / 64; ++it) {
    const int p = it & 1;
    const int m0 = it * 64;
    if (w == 0) __builtin_amdgcn_s_wait_tensorcnt(0);
    __syncthreads();  // publish buffer p; buffer p^1 free for next DMA
    if (w == 0 && it + 1 < NN / 64) {
      const int m1 = m0 + 64;
      tdm_load_2d((unsigned)(size_t)&klds[p ^ 1][0], Kb + (size_t)m1 * OC,
                  256u, 4096u, 256u, 64u, 256ull, 6u, 3u);
      tdm_load_2d((unsigned)(size_t)&vlds[p ^ 1][0], Vb + m1,
                  4096u, 256u, 64u, 256u, 4096ull, 4u, 3u);
    }

    // ---- S = Q K^T for 16 rows x 64 cols (K operand from LDS) ----------
    v8f s[4];
#pragma unroll
    for (int t = 0; t < 4; ++t) s[t] = vzero8();
#pragma unroll
    for (int j = 0; j < 8; ++j) {
      const bf16x16 qa = ld_g16(Qb + j * 32 + h * 16);
#pragma unroll
      for (int t = 0; t < 4; ++t) {
        const __bf16* kp = &klds[p][(t * 16 + l16) * KROW + j * 32 + h * 16];
        s[t] = wmma_bf16(qa, ld_g16(kp), s[t]);
      }
    }
#pragma unroll
    for (int t = 0; t < 4; ++t)
#pragma unroll
      for (int r = 0; r < 8; ++r) s[t][r] *= 0.0625f;  // 1/temp

    // ---- online softmax (rows live in 16-lane halves) ------------------
    float mnew[8], alpha[8], rs[8];
#pragma unroll
    for (int r = 0; r < 8; ++r) {
      float v = fmaxf(fmaxf(s[0][r], s[1][r]), fmaxf(s[2][r], s[3][r]));
      v = fmaxf(v, __shfl_xor(v, 1));
      v = fmaxf(v, __shfl_xor(v, 2));
      v = fmaxf(v, __shfl_xor(v, 4));
      v = fmaxf(v, __shfl_xor(v, 8));
      mnew[r]  = fmaxf(m_run[r], v);
      alpha[r] = __expf(m_run[r] - mnew[r]);
      m_run[r] = mnew[r];
      rs[r]    = 0.0f;
    }
#pragma unroll
    for (int t = 0; t < 4; ++t)
#pragma unroll
      for (int r = 0; r < 8; ++r) {
        const float pv = __expf(s[t][r] - mnew[r]);
        rs[r] += pv;
        pl[(r + 8 * h) * 72 + t * 16 + l16] = (__bf16)pv;  // D-layout -> row major
      }
#pragma unroll
    for (int r = 0; r < 8; ++r) {
      float v = rs[r];
      v += __shfl_xor(v, 1);
      v += __shfl_xor(v, 2);
      v += __shfl_xor(v, 4);
      v += __shfl_xor(v, 8);
      l_run[r] = l_run[r] * alpha[r] + v;
    }

    // ---- O = alpha*O + P V^T (V operand from LDS) ----------------------
    const bf16x16 pa0 = ld_g16(pl + l16 * 72 + h * 16);       // K = m 0..31
    const bf16x16 pa1 = ld_g16(pl + l16 * 72 + 32 + h * 16);  // K = m 32..63
#pragma unroll
    for (int ot = 0; ot < 16; ++ot) {
#pragma unroll
      for (int r = 0; r < 8; ++r) o_acc[ot][r] *= alpha[r];
      const __bf16* vp = &vlds[p][(ot * 16 + l16) * VROW + h * 16];
      o_acc[ot] = wmma_bf16(pa0, ld_g16(vp), o_acc[ot]);
      o_acc[ot] = wmma_bf16(pa1, ld_g16(vp + 32), o_acc[ot]);
    }
  }

  // ---- epilogue: out[b][o][n] = O[n][o] / l[n] -------------------------
#pragma unroll
  for (int ot = 0; ot < 16; ++ot)
#pragma unroll
    for (int r = 0; r < 8; ++r)
      out[((size_t)(b * OC) + ot * 16 + l16) * NN + n_base + r + 8 * h] =
          o_acc[ot][r] / l_run[r];
}

// ---------------------------------------------------------------------------
extern "C" void kernel_launch(void* const* d_in, const int* in_sizes, int n_in,
                              void* d_out, int out_size, void* d_ws,
                              size_t ws_size, hipStream_t stream) {
  const float* x  = (const float*)d_in[0];
  const float* xx = (const float*)d_in[1];
  const float* Wq = (const float*)d_in[2];
  const float* Wk = (const float*)d_in[3];
  const float* Wv = (const float*)d_in[4];
  float* out = (float*)d_out;

  const size_t elems = (size_t)BB * NN * OC;  // 4M elems, 8MB each as bf16
  __bf16* Q  = (__bf16*)d_ws;
  __bf16* K  = (__bf16*)((char*)d_ws + elems * sizeof(__bf16));
  __bf16* Vt = (__bf16*)((char*)d_ws + 2 * elems * sizeof(__bf16));

  dim3 grid1(NN / 128, BB, 3);
  proj_kernel<<<grid1, 256, 0, stream>>>(x, xx, Wq, Wk, Wv, Q, K, Vt);

  dim3 grid2(NN / 128, BB);
  attn_kernel<<<grid2, 256, 0, stream>>>(Q, K, Vt, out);
}
